// LSTM_57844619543162
// MI455X (gfx1250) — compile-verified, matrix operand order checked by reading, not run on hardware
//
#include <hip/hip_runtime.h>

// Problem constants (from reference)
#define B_ROWS  131072
#define IN_DIM  128
#define H_DIM   128
#define OUT_DIM 128
#define K_DIM   256          // IN + H
#define G4      512          // 4*H gate columns
#define M_TILE  64           // rows per chunk per workgroup
#define CHUNKS  4            // chunks per workgroup
#define GRID_WG (B_ROWS / M_TILE / CHUNKS)   // 512 workgroups

typedef __attribute__((ext_vector_type(16))) __bf16 v16bf;
typedef __attribute__((ext_vector_type(8)))  float  v8f;

union BF16x16 { v16bf v; unsigned int u[8]; };

// Truncate-pack two f32 into a bf16 pair with ONE v_perm_b32:
// D[15:0] = lo[31:16], D[31:16] = hi[31:16]
__device__ __forceinline__ unsigned int pack2_bf16(float lo, float hi) {
    return __builtin_amdgcn_perm(__float_as_uint(hi), __float_as_uint(lo), 0x07060302u);
}
__device__ __forceinline__ unsigned short trunc_bf16(float f) {
    return (unsigned short)(__float_as_uint(f) >> 16);
}

__device__ __forceinline__ float fast_sigmoid(float x) {
    return 1.0f / (1.0f + __expf(-x));
}
__device__ __forceinline__ float fast_tanh(float x) {
    x = fminf(15.0f, fmaxf(-15.0f, x));
    float e = __expf(2.0f * x);
    return (e - 1.0f) / (e + 1.0f);
}

// LDS layout (dynamic shared, 304 KB total <= 320 KB/WGP):
//   [0)        W4  bf16  [512][256]   (rows: 0-127 Wi, 128-255 Wf, 256-383 Wo, 384-511 Wg)
//   [256KB)    Wy  bf16  [128][128]
//   [288KB)    nh  bf16  [64][128]    per-chunk new_hidden staging
#define LDS_W4_OFF  0
#define LDS_WY_OFF  (G4 * K_DIM)
#define LDS_NH_OFF  (G4 * K_DIM + OUT_DIM * H_DIM)
#define LDS_USHORTS (G4 * K_DIM + OUT_DIM * H_DIM + M_TILE * H_DIM)

__global__ __launch_bounds__(256, 1) void lstm_fused_wmma(
    const float* __restrict__ x,   const float* __restrict__ hidden,
    const float* __restrict__ cell,
    const float* __restrict__ Wi,  const float* __restrict__ bi_,
    const float* __restrict__ Wf,  const float* __restrict__ bf_,
    const float* __restrict__ Wo,  const float* __restrict__ bo_,
    const float* __restrict__ Wg,  const float* __restrict__ bg_,
    const float* __restrict__ Wy,  const float* __restrict__ by_,
    float* __restrict__ out_y, float* __restrict__ out_h, float* __restrict__ out_c)
{
    extern __shared__ unsigned short smem[];
    unsigned short* lds_w4 = smem + LDS_W4_OFF;
    unsigned short* lds_wy = smem + LDS_WY_OFF;
    unsigned short* lds_nh = smem + LDS_NH_OFF;

    const int tid  = threadIdx.x;
    const int lane = tid & 31;
    const int wv   = tid >> 5;       // wave 0..7
    const int mw   = wv & 3;         // row-tile within 64-row chunk
    const int cw   = wv >> 2;        // h-column half (0: h[0,64), 1: h[64,128))
    const int ln   = lane & 15;      // N index within 16x16 tile
    const int kg   = lane >> 4;      // K-group (0/1) per WMMA operand layout

    // ---------------- stage weights f32 -> bf16 into LDS (once per WG) ----------------
    const float* wsrc[4] = { Wi, Wf, Wo, Wg };
    #pragma unroll 4
    for (int it = 0; it < (G4 * K_DIM / 4) / 256; ++it) {      // 128 iters, float4 each
        int e   = (tid + 256 * it) * 4;
        int row = e >> 8;                                       // gate column 0..511
        int k   = e & 255;
        const float4 f = *(const float4*)(wsrc[row >> 7] + (row & 127) * K_DIM + k);
        uint2 p;
        p.x = pack2_bf16(f.x, f.y);
        p.y = pack2_bf16(f.z, f.w);
        *(uint2*)(lds_w4 + row * K_DIM + k) = p;                // ds_store_b64
    }
    #pragma unroll 4
    for (int it = 0; it < (OUT_DIM * H_DIM / 4) / 256; ++it) {  // 16 iters
        int e = (tid + 256 * it) * 4;
        const float4 f = *(const float4*)(Wy + e);
        uint2 p;
        p.x = pack2_bf16(f.x, f.y);
        p.y = pack2_bf16(f.z, f.w);
        *(uint2*)(lds_wy + e) = p;
    }
    __syncthreads();

    for (int ci = 0; ci < CHUNKS; ++ci) {
        const int rowBlock = (blockIdx.x * CHUNKS + ci) * M_TILE;
        const int rowBase  = rowBlock + mw * 16;
        const int hBase    = cw * 64;

        // Prefetch next chunk's activations (one 128B line per thread per array):
        // emits global_prefetch_b8; hides HBM latency under this chunk's WMMAs.
        if (ci + 1 < CHUNKS) {
            const size_t nb = (size_t)(rowBlock + M_TILE) * H_DIM + tid * 32;
            __builtin_prefetch(x + nb, 0, 1);
            __builtin_prefetch(hidden + nb, 0, 1);
            __builtin_prefetch(cell + nb, 0, 1);
        }

        // ---------------- gate GEMM: [16 x 256] @ W4^T slice, bf16 WMMA ----------------
        v8f acc[4][4];                    // [gate i/f/o/g][h-subtile]
        #pragma unroll
        for (int g = 0; g < 4; ++g)
            #pragma unroll
            for (int t = 0; t < 4; ++t) acc[g][t] = (v8f)0.0f;

        #pragma unroll 2
        for (int kb = 0; kb < K_DIM / 32; ++kb) {
            // A fragment (16x32 bf16): lane ln = row, kg selects K half; 16 contiguous k
            const int ks   = kb * 32 + kg * 16;
            const int arow = rowBase + ln;
            const float* asrc = (ks < IN_DIM)
                ? (x      + (size_t)arow * IN_DIM + ks)
                : (hidden + (size_t)arow * H_DIM  + (ks - IN_DIM));
            BF16x16 a;
            #pragma unroll
            for (int q = 0; q < 4; ++q) {
                float4 f = ((const float4*)asrc)[q];
                a.u[2*q+0] = pack2_bf16(f.x, f.y);   // one v_perm_b32 each
                a.u[2*q+1] = pack2_bf16(f.z, f.w);
            }
            #pragma unroll
            for (int g = 0; g < 4; ++g) {
                #pragma unroll
                for (int t = 0; t < 4; ++t) {
                    // B fragment (32x16): lane ln = gate-column, 16 contiguous k from LDS
                    const int col = g * 128 + hBase + t * 16 + ln;
                    v16bf bfr = *(const v16bf*)(lds_w4 + col * K_DIM + kb * 32 + kg * 16);
                    acc[g][t] = __builtin_amdgcn_wmma_f32_16x16x32_bf16(
                        false, a.v, false, bfr, (short)0, acc[g][t], false, false);
                }
            }
        }

        // -------- epilogue: bias + sigmoid/tanh + cell update, stage nh in LDS --------
        #pragma unroll
        for (int t = 0; t < 4; ++t) {
            const int hcol = hBase + t * 16 + ln;
            const float vbi = bi_[hcol], vbf = bf_[hcol];
            const float vbo = bo_[hcol], vbg = bg_[hcol];
            #pragma unroll
            for (int j = 0; j < 8; ++j) {
                const int m = j + kg * 8;                 // C/D layout: VGPR j -> M=j / j+8
                const size_t idx = (size_t)(rowBase + m) * H_DIM + hcol;
                const float iv = fast_sigmoid(acc[0][t][j] + vbi);
                const float fv = fast_sigmoid(acc[1][t][j] + vbf);
                const float ov = fast_sigmoid(acc[2][t][j] + vbo);
                const float gv = fast_tanh   (acc[3][t][j] + vbg);
                const float cn = fv * cell[idx] + iv * gv;
                const float hn = ov * fast_tanh(cn);
                __builtin_nontemporal_store(cn, out_c + idx);
                __builtin_nontemporal_store(hn, out_h + idx);
                lds_nh[(mw * 16 + m) * H_DIM + hcol] = trunc_bf16(hn);
            }
        }
        __syncthreads();

        // ---------------- output GEMM: [16 x 128] @ Wy^T, bf16 WMMA ----------------
        v8f acc2[4];
        #pragma unroll
        for (int t = 0; t < 4; ++t) acc2[t] = (v8f)0.0f;

        #pragma unroll
        for (int kb = 0; kb < H_DIM / 32; ++kb) {
            const int ks = kb * 32 + kg * 16;
            v16bf a2 = *(const v16bf*)(lds_nh + (mw * 16 + ln) * H_DIM + ks);
            #pragma unroll
            for (int t = 0; t < 4; ++t) {
                const int ocol = cw * 64 + t * 16 + ln;
                v16bf b2 = *(const v16bf*)(lds_wy + ocol * H_DIM + ks);
                acc2[t] = __builtin_amdgcn_wmma_f32_16x16x32_bf16(
                    false, a2, false, b2, (short)0, acc2[t], false, false);
            }
        }
        #pragma unroll
        for (int t = 0; t < 4; ++t) {
            const int ocol = cw * 64 + t * 16 + ln;
            const float vby = by_[ocol];
            #pragma unroll
            for (int j = 0; j < 8; ++j) {
                const size_t idx = (size_t)(rowBase + j + kg * 8) * OUT_DIM + ocol;
                __builtin_nontemporal_store(acc2[t][j] + vby, out_y + idx);
            }
        }
        __syncthreads();   // protect lds_nh before next chunk overwrites it
    }
}

extern "C" void kernel_launch(void* const* d_in, const int* in_sizes, int n_in,
                              void* d_out, int out_size, void* d_ws, size_t ws_size,
                              hipStream_t stream) {
    const float* x      = (const float*)d_in[0];
    const float* hidden = (const float*)d_in[1];
    const float* cell   = (const float*)d_in[2];
    const float* Wi     = (const float*)d_in[3];
    const float* bi     = (const float*)d_in[4];
    const float* Wf     = (const float*)d_in[5];
    const float* bf     = (const float*)d_in[6];
    const float* Wo     = (const float*)d_in[7];
    const float* bo     = (const float*)d_in[8];
    const float* Wg     = (const float*)d_in[9];
    const float* bg     = (const float*)d_in[10];
    const float* Wy     = (const float*)d_in[11];
    const float* by     = (const float*)d_in[12];

    float* out_y = (float*)d_out;                       // output      [B, OUT]
    float* out_h = out_y + (size_t)B_ROWS * OUT_DIM;    // new_hidden  [B, H]
    float* out_c = out_h + (size_t)B_ROWS * H_DIM;      // new_cell    [B, H]

    const size_t lds_bytes = (size_t)LDS_USHORTS * sizeof(unsigned short); // 304 KB
    (void)hipFuncSetAttribute(reinterpret_cast<const void*>(lstm_fused_wmma),
                              hipFuncAttributeMaxDynamicSharedMemorySize,
                              (int)lds_bytes);

    lstm_fused_wmma<<<GRID_WG, 256, lds_bytes, stream>>>(
        x, hidden, cell, Wi, bi, Wf, bf, Wo, bo, Wg, bg, Wy, by,
        out_y, out_h, out_c);
}
